// MultiHierarchicalBiGRU_11690900980384
// MI455X (gfx1250) — matrix operand order repaired and checked by reading, not compiled, not run on hardware
//
#include <hip/hip_runtime.h>
#include <hip/hip_bf16.h>
#include <cstdint>
#include <cstddef>

// ---------------------------------------------------------------------------
// Types for CDNA5 WMMA (wave32): v_wmma_f32_16x16x32_bf16
// ---------------------------------------------------------------------------
typedef __attribute__((ext_vector_type(16))) __bf16          bf16x16;
typedef __attribute__((ext_vector_type(8)))  float           f32x8;
typedef __attribute__((ext_vector_type(8)))  unsigned short  u16x8;
typedef __attribute__((ext_vector_type(16))) unsigned short  u16x16;

// Model dims
#define E_DIM 256
#define H_DIM 256
#define G3    768     // 3*H
#define NW    2048    // B*R*S word-level batch
#define NS    128     // B*R sentence-level batch
#define NB    16      // review-level batch padded (8 valid)
#define ROWS  16      // batch rows owned by one workgroup (one M-tile)

__device__ __forceinline__ unsigned short f32_to_bf16_rne(float f) {
    unsigned u = __builtin_bit_cast(unsigned, f);
    unsigned r = u + 0x7FFFu + ((u >> 16) & 1u);
    return (unsigned short)(r >> 16);
}

__device__ __forceinline__ float sigmoidf(float x) {
    return 1.0f / (1.0f + expf(-x));
}

// Build a 16-element bf16 fragment from two contiguous 8-element (16B) chunks.
__device__ __forceinline__ bf16x16 frag_pair(const unsigned short* p0,
                                             const unsigned short* p1) {
    u16x8 a = *reinterpret_cast<const u16x8*>(p0);
    u16x8 b = *reinterpret_cast<const u16x8*>(p1);
    u16x16 c = __builtin_shufflevector(a, b, 0,1,2,3,4,5,6,7,8,9,10,11,12,13,14,15);
    return __builtin_bit_cast(bf16x16, c);
}

// ---------------------------------------------------------------------------
// Fused biGRU level. One workgroup (8 waves) owns ROWS=16 batch rows and runs
// all T steps with h resident in LDS (rows of h_{t+1} depend only on the same
// rows of h_t, so the recurrence is private to the block). Per step each wave
// owns two hidden 16x16 j-tiles, processed one at a time with 6 fp32 WMMA
// accumulators ({gx,gh} x {r,z,n}). Per K-block, all six B fragments are
// fetched into registers *before* the six WMMAs are issued, so 12 b128 loads
// are in flight per wait and the unrolled K loop can overlap the next block's
// loads with this block's WMMAs. An asm memory barrier at the top of each
// step keeps LICM from hoisting the weight loads (-> no scratch spilling).
//   x      : [T][Npad][256] bf16 (time-major)
//   wih/whh: [2][768][256]  bf16 (PyTorch row-major == WMMA-B friendly)
//   bih/bhh: [2][768]       f32
//   hout   : [2][Npad][256] f32 (final hidden per direction)
// blockIdx.x = row block, blockIdx.y = direction (1 => reversed time)
// ---------------------------------------------------------------------------
__global__ __launch_bounds__(256) void gru_level_kernel(
    const unsigned short* __restrict__ x,
    const unsigned short* __restrict__ wih,
    const unsigned short* __restrict__ whh,
    const float* __restrict__ bih2,
    const float* __restrict__ bhh2,
    float* __restrict__ hout,
    int T, int Npad)
{
    __shared__ float          hf[ROWS][H_DIM];     // fp32 hidden state (16 KB)
    __shared__ unsigned short hb[ROWS][H_DIM];     // bf16 snapshot     (8 KB)

    const int dir     = blockIdx.y;
    const int rowbase = blockIdx.x * ROWS;
    const unsigned short* wih_d = wih + (size_t)dir * (G3 * H_DIM);
    const unsigned short* whh_d = whh + (size_t)dir * (G3 * H_DIM);
    const float* bih_d = bih2 + dir * G3;
    const float* bhh_d = bhh2 + dir * G3;

    const int tid  = threadIdx.x;
    const int lane = tid & 31;
    const int wave = tid >> 5;
    const int l15  = lane & 15;      // row-in-tile (A/C) / col-in-tile (B/C)
    const int lh   = lane >> 4;      // lane half selects K / M sub-block
    const int jt0  = wave * 2;       // first of this wave's two j-tiles

    // Gate biases for this wave's two columns (constant across time).
    float bias_r[2], bias_z[2], bias_ni[2], bias_nh[2];
#pragma unroll
    for (int u = 0; u < 2; ++u) {
        const int jcol = (jt0 + u) * 16 + l15;
        bias_r[u]  = bih_d[jcol]             + bhh_d[jcol];
        bias_z[u]  = bih_d[H_DIM + jcol]     + bhh_d[H_DIM + jcol];
        bias_ni[u] = bih_d[2 * H_DIM + jcol];
        bias_nh[u] = bhh_d[2 * H_DIM + jcol];
    }
    const unsigned short* ahp = &hb[l15][0] + lh * 8;   // A_h lane row base
    const int m0 = lh * 8;

    float* hff = &hf[0][0];
    for (int i = tid; i < ROWS * H_DIM; i += 256) hff[i] = 0.0f;
    __syncthreads();

    for (int t = 0; t < T; ++t) {
        // Keep weight-fragment loads inside the time loop (no LICM -> no
        // scratch spilling); they hit WGP$/L2 every step instead.
        asm volatile("" ::: "memory");

        // Snapshot h -> bf16 for this step's A fragments.
        unsigned short* hbf = &hb[0][0];
        for (int i = tid; i < ROWS * H_DIM; i += 256) hbf[i] = f32_to_bf16_rne(hff[i]);
        // Prefetch next timestep's x tile (8 KB) -> global_prefetch_b8.
        if (t + 1 < T) {
            const int txn = dir ? (T - 2 - t) : (t + 1);
            const unsigned short* nx =
                x + ((size_t)txn * Npad + rowbase) * E_DIM + (size_t)tid * 16;
            __builtin_prefetch(nx, 0, 0);
        }
        __syncthreads();

        const int tx = dir ? (T - 1 - t) : t;
        const unsigned short* xt = x + ((size_t)tx * Npad + rowbase) * E_DIM;
        const unsigned short* axp = xt + ((size_t)l15) * E_DIM + lh * 8;

        // One j-tile at a time: 6 live accumulators, fragments streamed in K.
#pragma clang loop unroll(disable)
        for (int u = 0; u < 2; ++u) {
            const int jcol = (jt0 + u) * 16 + l15;
            const unsigned short* bip0 = wih_d + (size_t)jcol * H_DIM + lh * 8;
            const unsigned short* bhp0 = whh_d + (size_t)jcol * H_DIM + lh * 8;

            f32x8 acc[6] = {};   // gx_r, gx_z, gx_n, gh_r, gh_z, gh_n
#pragma unroll
            for (int kb8 = 0; kb8 < 8; ++kb8) {
                const int kb = kb8 * 32;
                // Batch all loads of this K-block first: 2 A frags + 6 B frags
                // (12x b128 for B) in flight before any WMMA waits on them.
                bf16x16 ax = frag_pair(axp + kb, axp + kb + 16);
                bf16x16 ah = frag_pair(ahp + kb, ahp + kb + 16);
                bf16x16 bx[3], bh[3];
#pragma unroll
                for (int g = 0; g < 3; ++g) {
                    const size_t go = (size_t)g * (H_DIM * H_DIM);
                    bx[g] = frag_pair(bip0 + go + kb, bip0 + go + kb + 16);
                    bh[g] = frag_pair(bhp0 + go + kb, bhp0 + go + kb + 16);
                }
                // Six WMMAs rotating through six accumulators (no acc chaining).
#pragma unroll
                for (int g = 0; g < 3; ++g) {
                    acc[g] = __builtin_amdgcn_wmma_f32_16x16x32_bf16(
                        false, ax, false, bx[g], (short)0, acc[g], false, false);
                    acc[3 + g] = __builtin_amdgcn_wmma_f32_16x16x32_bf16(
                        false, ah, false, bh[g], (short)0, acc[3 + g], false, false);
                }
            }

            // Gate math in C-fragment layout: lane = column jcol, rows m0..m0+7.
#pragma unroll
            for (int v = 0; v < 8; ++v) {
                float hp = hf[m0 + v][jcol];
                float rg = sigmoidf(acc[0][v] + acc[3][v] + bias_r[u]);
                float zg = sigmoidf(acc[1][v] + acc[4][v] + bias_z[u]);
                float ng = tanhf(acc[2][v] + bias_ni[u] +
                                 rg * (acc[5][v] + bias_nh[u]));
                hf[m0 + v][jcol] = (1.0f - zg) * ng + zg * hp;
            }
        }
        __syncthreads();
    }

    float* outp = hout + ((size_t)dir * Npad + rowbase) * H_DIM;
    for (int i = tid; i < ROWS * H_DIM; i += 256) outp[i] = hff[i];
}

// ---------------------------------------------------------------------------
// Small helper kernels
// ---------------------------------------------------------------------------
__global__ void cvt_bf16_kernel(const float* __restrict__ src,
                                unsigned short* __restrict__ dst, int count) {
    int i = blockIdx.x * 256 + threadIdx.x;
    if (i < count) dst[i] = f32_to_bf16_rne(src[i]);
}

// tokens [2048 sentences][32 words] -> x1 bf16 time-major [32][2048][256]
__global__ void embed_kernel(const int* __restrict__ tok,
                             const float* __restrict__ emb,
                             unsigned short* __restrict__ x1) {
    int id = blockIdx.x;               // n*32 + t
    int n = id >> 5, t = id & 31;
    int e = threadIdx.x;
    int token = tok[id];
    float v = emb[(size_t)token * E_DIM + e];
    x1[((size_t)t * NW + n) * E_DIM + e] = f32_to_bf16_rne(v);
}

// sent [2][2048][256] f32 -> x2 bf16 [16(s)][128(br)][256]; n = br*16+s
__global__ void combine12_kernel(const float* __restrict__ sent,
                                 unsigned short* __restrict__ x2) {
    int n = blockIdx.x;
    int e = threadIdx.x;
    int s = n & 15, br = n >> 4;
    float v = sent[(size_t)n * H_DIM + e] + sent[(size_t)(NW + n) * H_DIM + e];
    x2[((size_t)(s * NS + br)) * H_DIM + e] = f32_to_bf16_rne(v);
}

// rev [2][128][256] f32 -> x3 bf16 [16(r)][16(b,pad)][256]; n = b*16+r
__global__ void combine23_kernel(const float* __restrict__ rev,
                                 unsigned short* __restrict__ x3) {
    int id = blockIdx.x;               // r*16 + bb
    int r = id >> 4, bb = id & 15;
    int e = threadIdx.x;
    float v = 0.0f;
    if (bb < 8) {
        int n = bb * 16 + r;
        v = rev[(size_t)n * H_DIM + e] + rev[(size_t)(NS + n) * H_DIM + e];
    }
    x3[(size_t)id * H_DIM + e] = f32_to_bf16_rne(v);
}

// Linear(256->128) -> SELU -> Linear(128->1); input = inA[n] + inB[n]
__global__ __launch_bounds__(128) void fc_head_kernel(
    const float* __restrict__ inA, const float* __restrict__ inB,
    const float* __restrict__ W1, const float* __restrict__ b1,
    const float* __restrict__ W2, const float* __restrict__ b2,
    float* __restrict__ out) {
    __shared__ float xs[256];
    __shared__ float red[128];
    int n = blockIdx.x, j = threadIdx.x;
    xs[j]       = inA[(size_t)n * 256 + j]       + inB[(size_t)n * 256 + j];
    xs[j + 128] = inA[(size_t)n * 256 + j + 128] + inB[(size_t)n * 256 + j + 128];
    __syncthreads();
    float s = b1[j];
#pragma unroll 8
    for (int e = 0; e < 256; ++e) s += xs[e] * W1[(size_t)j * 256 + e];
    const float alpha = 1.6732632423543772f, scale = 1.0507009873554805f;
    float t1 = scale * (s > 0.0f ? s : alpha * (expf(s) - 1.0f));
    red[j] = t1 * W2[j];
    __syncthreads();
    for (int st = 64; st > 0; st >>= 1) {
        if (j < st) red[j] += red[j + st];
        __syncthreads();
    }
    if (j == 0) out[n] = red[0] + b2[0];
}

// ---------------------------------------------------------------------------
// Launcher
// ---------------------------------------------------------------------------
extern "C" void kernel_launch(void* const* d_in, const int* in_sizes, int n_in,
                              void* d_out, int out_size, void* d_ws, size_t ws_size,
                              hipStream_t stream) {
    (void)in_sizes; (void)n_in; (void)out_size; (void)ws_size;

    const int*   tok    = (const int*)  d_in[0];
    const float* emb    = (const float*)d_in[1];
    const float* w_Wih  = (const float*)d_in[2];
    const float* w_Whh  = (const float*)d_in[3];
    const float* w_bih  = (const float*)d_in[4];
    const float* w_bhh  = (const float*)d_in[5];
    const float* s_Wih  = (const float*)d_in[6];
    const float* s_Whh  = (const float*)d_in[7];
    const float* s_bih  = (const float*)d_in[8];
    const float* s_bhh  = (const float*)d_in[9];
    const float* r_Wih  = (const float*)d_in[10];
    const float* r_Whh  = (const float*)d_in[11];
    const float* r_bih  = (const float*)d_in[12];
    const float* r_bhh  = (const float*)d_in[13];
    const float* rfc_W1 = (const float*)d_in[14];
    const float* rfc_b1 = (const float*)d_in[15];
    const float* rfc_W2 = (const float*)d_in[16];
    const float* rfc_b2 = (const float*)d_in[17];
    const float* pfc_W1 = (const float*)d_in[18];
    const float* pfc_b1 = (const float*)d_in[19];
    const float* pfc_W2 = (const float*)d_in[20];
    const float* pfc_b2 = (const float*)d_in[21];
    float* out = (float*)d_out;

    // Workspace carve-up (all 256B-aligned sizes)
    char* ws = (char*)d_ws;
    size_t off = 0;
    auto take = [&](size_t bytes) { char* p = ws + off; off += (bytes + 255) & ~(size_t)255; return p; };

    unsigned short* x1   = (unsigned short*)take((size_t)32 * NW * E_DIM * 2);  // 32 MB
    unsigned short* wihw = (unsigned short*)take((size_t)2 * G3 * H_DIM * 2);
    unsigned short* whhw = (unsigned short*)take((size_t)2 * G3 * H_DIM * 2);
    unsigned short* wihs = (unsigned short*)take((size_t)2 * G3 * H_DIM * 2);
    unsigned short* whhs = (unsigned short*)take((size_t)2 * G3 * H_DIM * 2);
    unsigned short* wihr = (unsigned short*)take((size_t)2 * G3 * H_DIM * 2);
    unsigned short* whhr = (unsigned short*)take((size_t)2 * G3 * H_DIM * 2);
    float*          sent = (float*)take((size_t)2 * NW * H_DIM * 4);
    unsigned short* x2   = (unsigned short*)take((size_t)16 * NS * H_DIM * 2);
    float*          rev  = (float*)take((size_t)2 * NS * H_DIM * 4);
    unsigned short* x3   = (unsigned short*)take((size_t)16 * NB * H_DIM * 2);
    float*          biz  = (float*)take((size_t)2 * NB * H_DIM * 4);

    const int wcount = 2 * G3 * H_DIM;           // 393216
    const int wgrid  = (wcount + 255) / 256;
    cvt_bf16_kernel<<<wgrid, 256, 0, stream>>>(w_Wih, wihw, wcount);
    cvt_bf16_kernel<<<wgrid, 256, 0, stream>>>(w_Whh, whhw, wcount);
    cvt_bf16_kernel<<<wgrid, 256, 0, stream>>>(s_Wih, wihs, wcount);
    cvt_bf16_kernel<<<wgrid, 256, 0, stream>>>(s_Whh, whhs, wcount);
    cvt_bf16_kernel<<<wgrid, 256, 0, stream>>>(r_Wih, wihr, wcount);
    cvt_bf16_kernel<<<wgrid, 256, 0, stream>>>(r_Whh, whhr, wcount);

    embed_kernel<<<NW * 32, 256, 0, stream>>>(tok, emb, x1);

    // Word-level biGRU: 2048 rows, T=32 -> 128 x 2 workgroups
    gru_level_kernel<<<dim3(NW / ROWS, 2), 256, 0, stream>>>(
        x1, wihw, whhw, w_bih, w_bhh, sent, 32, NW);
    combine12_kernel<<<NW, 256, 0, stream>>>(sent, x2);

    // Sentence-level biGRU: 128 rows, T=16 -> 8 x 2 workgroups
    gru_level_kernel<<<dim3(NS / ROWS, 2), 256, 0, stream>>>(
        x2, wihs, whhs, s_bih, s_bhh, rev, 16, NS);
    combine23_kernel<<<16 * NB, 256, 0, stream>>>(rev, x3);

    // Review-level biGRU: 8 rows padded to 16, T=16 -> 1 x 2 workgroups
    gru_level_kernel<<<dim3(1, 2), 256, 0, stream>>>(
        x3, wihr, whhr, r_bih, r_bhh, biz, 16, NB);

    // Heads: b_stars -> out[0..7], r_stars -> out[8..135]
    fc_head_kernel<<<NS, 128, 0, stream>>>(rev, rev + (size_t)NS * H_DIM,
                                           rfc_W1, rfc_b1, rfc_W2, rfc_b2, out + 8);
    fc_head_kernel<<<8, 128, 0, stream>>>(biz, biz + (size_t)NB * H_DIM,
                                          pfc_W1, pfc_b1, pfc_W2, pfc_b2, out);
}
